// CrossAttention_45578192945915
// MI455X (gfx1250) — matrix operand order repaired and checked by reading, not compiled, not run on hardware
//
#include <hip/hip_runtime.h>

// ---------------------------------------------------------------------------
// CDNA5 (gfx1250) cross-attention: conv3d(q,k,v) as shifted bf16 WMMA GEMMs,
// then attention (scores GEMM -> softmax -> out GEMM) + residual.
// Wave32, V_WMMA_F32_16X16X32_BF16; double-buffered LDS fed by
// GLOBAL_LOAD_ASYNC_TO_LDS_B128 (ASYNCcnt) per CDNA5 ISA.
// Wave tiles are 32x64 (8 WMMAs per barrier) for issue/barrier amortization.
// ---------------------------------------------------------------------------

typedef __attribute__((ext_vector_type(16))) __bf16         bf16x16;
typedef __attribute__((ext_vector_type(8)))  float          f32x8;
typedef __attribute__((ext_vector_type(8)))  unsigned short u16x8;

static constexpr int BATCH = 16;
static constexpr int CCH   = 1024;   // channels
static constexpr int NTOK  = 512;    // D*H*W tokens

#define USE_ASYNC 1

// float -> bf16 round-to-nearest-even (pure integer, always compiles)
static __device__ __forceinline__ unsigned short f2bf(float f) {
    unsigned u = __builtin_bit_cast(unsigned, f);
    u += 0x7fffu + ((u >> 16) & 1u);
    return (unsigned short)(u >> 16);
}

struct U16x8Pair { u16x8 a, b; };

// 16B global -> LDS stage. Async path: per-lane async DMA, no VGPR round-trip.
static __device__ __forceinline__ void stage16(unsigned short* l,
                                               const unsigned short* g) {
#if USE_ASYNC
    // LDS byte address comes from the vdst VGPR; low 32 bits of a generic LDS pointer
    // are the wave-relative LDS byte offset.
    asm volatile("global_load_async_to_lds_b128 %0, %1, off"
                 :: "v"((unsigned)(uintptr_t)(void*)l), "v"(g)
                 : "memory");
#else
    *(uint4*)l = *(const uint4*)g;
#endif
}

// Must precede the barrier that publishes async-staged LDS data.
static __device__ __forceinline__ void async_fence() {
#if USE_ASYNC
    asm volatile("s_wait_asynccnt 0" ::: "memory");
#endif
}

// Load a 16-element bf16 fragment as two 16B LDS reads (ds_load_b128 x2)
static __device__ __forceinline__ bf16x16 lds_frag(const unsigned short* p0,
                                                   const unsigned short* p1) {
    U16x8Pair t;
    t.a = *(const u16x8*)p0;
    t.b = *(const u16x8*)p1;
    return __builtin_bit_cast(bf16x16, t);
}

static __device__ __forceinline__ f32x8 wmma_bf16(bf16x16 a, bf16x16 b, f32x8 c) {
    // (neg_a, A, neg_b, B, c_mod, C, reuse_a, reuse_b)
    return __builtin_amdgcn_wmma_f32_16x16x32_bf16(false, a, false, b,
                                                   (short)0, c, false, false);
}

// ---------------------------------------------------------------------------
// Kernel 1: repack weights fp32 [co][ci][27] -> bf16 wpk[t][co][ci]
// ---------------------------------------------------------------------------
__global__ __launch_bounds__(256) void pack_weights_kernel(
    const float* __restrict__ wq, const float* __restrict__ wk,
    const float* __restrict__ wv,
    unsigned short* __restrict__ pq, unsigned short* __restrict__ pk,
    unsigned short* __restrict__ pv) {
    const int sel = blockIdx.y;
    const float* src = (sel == 0) ? wq : (sel == 1) ? wk : wv;
    unsigned short* dst = (sel == 0) ? pq : (sel == 1) ? pk : pv;

    const size_t idx = (size_t)blockIdx.x * 256 + threadIdx.x; // < 27*2^20
    const int ci = (int)(idx & 1023);
    const int co = (int)((idx >> 10) & 1023);
    const int t  = (int)(idx >> 20);
    dst[idx] = f2bf(src[((size_t)co * CCH + ci) * 27 + t]);
}

// ---------------------------------------------------------------------------
// Kernel 2: transpose+convert activations fp32 [b][c][n] -> bf16 [b][n][c]
// ---------------------------------------------------------------------------
__global__ __launch_bounds__(256) void transpose_in_kernel(
    const float* __restrict__ x, const float* __restrict__ y,
    unsigned short* __restrict__ xt, unsigned short* __restrict__ yt) {
    __shared__ float tile[32][33];
    const int z = blockIdx.z;
    const int b = z >> 1;
    const float* src = (z & 1) ? y : x;
    unsigned short* dst = (z & 1) ? yt : xt;
    const int nT = blockIdx.x * 32, cT = blockIdx.y * 32;
    const int tx = threadIdx.x & 31, ty = threadIdx.x >> 5;
#pragma unroll
    for (int i = 0; i < 4; ++i)
        tile[ty + i * 8][tx] =
            src[((size_t)b * CCH + cT + ty + i * 8) * NTOK + nT + tx];
    __syncthreads();
#pragma unroll
    for (int i = 0; i < 4; ++i)
        dst[((size_t)b * NTOK + nT + ty + i * 8) * CCH + cT + tx] =
            f2bf(tile[tx][ty + i * 8]);
}

// ---------------------------------------------------------------------------
// Kernel 3: conv3d as implicit GEMM (27 shifted GEMMs folded into one K loop
// of 864 32-deep steps).  WG: 8 waves, tile M=64 tokens x N=256 ch;
// wave tile 32x64 -> 8 WMMAs per k-step.  Double-buffered LDS, async staging.
// Output: token-major bf16 (q,k) or channel-major bf16 (v) via chanMajor.
// ---------------------------------------------------------------------------
#define LP 40  // LDS row pitch (bf16 elems): 80B rows, banks well spread

__global__ __launch_bounds__(256) void conv_gemm_kernel(
    const unsigned short* __restrict__ srcT,  // [b][n][ci] bf16
    const unsigned short* __restrict__ wpk,   // [t][co][ci] bf16
    const float* __restrict__ bias,
    unsigned short* __restrict__ outT,        // see chanMajor
    const int chanMajor) {
    __shared__ __align__(16) unsigned short As[2 * 64 * LP];   // 10,240 B
    __shared__ __align__(16) unsigned short Bs[2 * 256 * LP];  // 40,960 B

    const int tid  = threadIdx.x;
    const int lane = tid & 31, wid = tid >> 5;
    const int l15 = lane & 15, hl = lane >> 4;
    const int waveM = wid >> 2, waveN = wid & 3;        // 2 x 4 wave grid
    const int b = blockIdx.z;
    const int dTok = blockIdx.y;                        // M-tile == depth slice

    // A staging: 1 chunk/thread (64 rows x 4 x 16B)
    const int am = tid >> 2, akp = tid & 3;
    const int hTok = am >> 3, wTok = am & 7;

    const f32x8 zero8 = {0.f, 0.f, 0.f, 0.f, 0.f, 0.f, 0.f, 0.f};
    f32x8 acc[2][4];
#pragma unroll
    for (int i = 0; i < 2; ++i)
#pragma unroll
        for (int j = 0; j < 4; ++j) acc[i][j] = zero8;

    auto stageC = [&](int s, int buf) {
        const int t = s >> 5, k0 = (s & 31) * 32;
        const int kd = t / 9, kh = (t % 9) / 3, kw = t % 3;
        const int dd = dTok + kd - 1, hh = hTok + kh - 1, ww = wTok + kw - 1;
        const bool valid = ((unsigned)dd < 8u) & ((unsigned)hh < 8u) &
                           ((unsigned)ww < 8u);
        unsigned short* Ab = As + buf * (64 * LP);
        unsigned short* Bb = Bs + buf * (256 * LP);
        if (valid) {
            stage16(Ab + am * LP + akp * 8,
                    srcT + ((size_t)b * NTOK + (dd * 64 + hh * 8 + ww)) * CCH +
                        k0 + akp * 8);
        } else {
            *(uint4*)(Ab + am * LP + akp * 8) = make_uint4(0u, 0u, 0u, 0u);
        }
        const size_t wTap = (size_t)t * CCH * CCH;
#pragma unroll
        for (int c2 = 0; c2 < 4; ++c2) {                // 256 rows x 4 chunks
            const int ch = tid + c2 * 256;
            const int co = ch >> 2, kp = ch & 3;
            stage16(Bb + co * LP + kp * 8,
                    wpk + wTap + ((size_t)blockIdx.x * 256 + co) * CCH + k0 +
                        kp * 8);
        }
        // prefetch the following weight k-slab into GL2
        __builtin_prefetch(wpk + wTap +
                               ((size_t)blockIdx.x * 256 + tid) * CCH +
                               ((k0 + 32) & 1023),
                           0, 1);
    };

    stageC(0, 0);
    for (int s = 0; s < 27 * 32; ++s) {
        const int buf = s & 1;
        async_fence();
        __syncthreads();
        if (s + 1 < 27 * 32) stageC(s + 1, buf ^ 1);

        const unsigned short* Ab = As + buf * (64 * LP);
        const unsigned short* Bb = Bs + buf * (256 * LP);
        bf16x16 aF[2], bF[4];
#pragma unroll
        for (int i = 0; i < 2; ++i) {
            const unsigned short* p =
                Ab + (waveM * 32 + i * 16 + l15) * LP + 8 * hl;
            aF[i] = lds_frag(p, p + 16);
        }
#pragma unroll
        for (int j = 0; j < 4; ++j) {
            const unsigned short* p =
                Bb + (waveN * 64 + j * 16 + l15) * LP + 8 * hl;
            bF[j] = lds_frag(p, p + 16);
        }
#pragma unroll
        for (int i = 0; i < 2; ++i)
#pragma unroll
            for (int j = 0; j < 4; ++j)
                acc[i][j] = wmma_bf16(aF[i], bF[j], acc[i][j]);
    }

    // ---- epilogue: +bias, bf16 store
#pragma unroll
    for (int j = 0; j < 4; ++j) {
        const int colG = blockIdx.x * 256 + waveN * 64 + j * 16 + l15;
        const float bv = bias[colG];
#pragma unroll
        for (int i = 0; i < 2; ++i) {
            const int tokBase = dTok * 64 + waveM * 32 + i * 16 + 8 * hl;
            if (chanMajor) {
                // [b][co][n]: 8 consecutive tokens -> one 16B store
                u16x8 v;
#pragma unroll
                for (int r = 0; r < 8; ++r) v[r] = f2bf(acc[i][j][r] + bv);
                *(u16x8*)(outT + ((size_t)b * CCH + colG) * NTOK + tokBase) = v;
            } else {
                // [b][n][co]
#pragma unroll
                for (int r = 0; r < 8; ++r)
                    outT[((size_t)b * NTOK + tokBase + r) * CCH + colG] =
                        f2bf(acc[i][j][r] + bv);
            }
        }
    }
}

// ---------------------------------------------------------------------------
// Kernel 4: attention for one (batch, 32-token tile).
//   scores = q.k (M=32,N=512,K=1024) -> softmax -> out = P.v (M=32,N=1024,K=512)
//   epilogue: out[b][c][n] = O[n][c] + x[b][c][n]
// Dynamic LDS 215,168 B (legal under CDNA5 320KB/WGP); Bs double-buffered.
// Wave tiles 32x64 in both GEMMs (8 WMMAs per barrier).
// ---------------------------------------------------------------------------
static constexpr int AQ_PITCH = 1032;              // bf16 elems
static constexpr int S_PITCH  = 516;               // f32 elems
static constexpr int AP_PITCH = 520;               // bf16 elems
static constexpr size_t AQ_OFF   = 0;
static constexpr size_t S_OFF    = 66048;          // 32*1032*2
static constexpr size_t BS_OFF   = 132096;         // + 32*516*4
static constexpr size_t BS_STRIDE = 40960;         // 512*LP*2 per buffer
static constexpr size_t RED_OFF  = 214016;         // + 2*40960
static constexpr size_t REDF_OFF = 215040;         // + 32*8*4
static constexpr size_t ATT_SMEM = 215168;         // + 32*4

__global__ __launch_bounds__(256) void attention_kernel(
    const unsigned short* __restrict__ qt,  // [b][n][c] bf16
    const unsigned short* __restrict__ kt,  // [b][n][c] bf16
    const unsigned short* __restrict__ vc,  // [b][c][n] bf16 (channel-major!)
    const float* __restrict__ x,            // [b][c][n] f32 (residual)
    float* __restrict__ out) {              // [b][c][n] f32
    extern __shared__ __align__(16) char smem[];
    unsigned short* Aq  = (unsigned short*)(smem + AQ_OFF);
    float*          S   = (float*)(smem + S_OFF);
    unsigned short* Ap  = (unsigned short*)(smem + AQ_OFF);  // aliases Aq
    float*          red  = (float*)(smem + RED_OFF);
    float*          redF = (float*)(smem + REDF_OFF);

    const int tid = threadIdx.x;
    const int lane = tid & 31, wid = tid >> 5;
    const int l15 = lane & 15, hl = lane >> 4;
    const int b = blockIdx.y;
    const int n0 = blockIdx.x * 32;                 // query-token base

    auto BsBuf = [&](int buf) {
        return (unsigned short*)(smem + BS_OFF + (size_t)buf * BS_STRIDE);
    };

    // ---- Phase 0: stage Aq = qt rows (32 x 1024 bf16), one-time
#pragma unroll
    for (int it = 0; it < 16; ++it) {
        const int ch = tid + it * 256;              // 4096 chunks
        const int row = ch >> 7, kp = ch & 127;
        *(uint4*)(Aq + row * AQ_PITCH + kp * 8) =
            *(const uint4*)(qt + ((size_t)b * NTOK + n0 + row) * CCH + kp * 8);
    }

    const f32x8 zero8 = {0.f, 0.f, 0.f, 0.f, 0.f, 0.f, 0.f, 0.f};

    // ---- Phase 1: scores GEMM. Each wave: 32 rows x 64 cols -> 2x4 frags.
    {
        auto stageS = [&](int kc, int buf) {
            const int k0 = kc * 32;
            unsigned short* Bb = BsBuf(buf);
#pragma unroll
            for (int it = 0; it < 8; ++it) {        // 512 rows x 4 chunks
                const int ch = tid + it * 256;
                const int m = ch >> 2, kp = ch & 3;
                stage16(Bb + m * LP + kp * 8,
                        kt + ((size_t)b * NTOK + m) * CCH + k0 + kp * 8);
            }
        };

        f32x8 accS[2][4];
#pragma unroll
        for (int i = 0; i < 2; ++i)
#pragma unroll
            for (int j = 0; j < 4; ++j) accS[i][j] = zero8;

        stageS(0, 0);
        for (int kc = 0; kc < 32; ++kc) {
            const int buf = kc & 1;
            const int k0 = kc * 32;
            async_fence();
            __syncthreads();
            if (kc + 1 < 32) stageS(kc + 1, buf ^ 1);

            const unsigned short* Bb = BsBuf(buf);
            bf16x16 aF[2], bF[4];
#pragma unroll
            for (int i = 0; i < 2; ++i) {
                const unsigned short* p =
                    Aq + (i * 16 + l15) * AQ_PITCH + k0 + 8 * hl;
                aF[i] = lds_frag(p, p + 16);
            }
#pragma unroll
            for (int j = 0; j < 4; ++j) {
                const unsigned short* p =
                    Bb + (wid * 64 + j * 16 + l15) * LP + 8 * hl;
                bF[j] = lds_frag(p, p + 16);
            }
#pragma unroll
            for (int i = 0; i < 2; ++i)
#pragma unroll
                for (int j = 0; j < 4; ++j)
                    accS[i][j] = wmma_bf16(aF[i], bF[j], accS[i][j]);
        }
        __syncthreads();
#pragma unroll
        for (int i = 0; i < 2; ++i)
#pragma unroll
            for (int j = 0; j < 4; ++j) {
                const int m = wid * 64 + j * 16 + l15;
#pragma unroll
                for (int r = 0; r < 8; ++r)
                    S[(i * 16 + 8 * hl + r) * S_PITCH + m] = accS[i][j][r];
            }
        __syncthreads();
    }

    // ---- Phase 2: row softmax over 512 cols (8 threads per row)
    {
        const int r = tid >> 3, l8 = tid & 7;
        float mx = -3.0e38f;
        for (int c = l8; c < 512; c += 8) mx = fmaxf(mx, S[r * S_PITCH + c]);
        red[r * 8 + l8] = mx;
        __syncthreads();
        if (tid < 32) {
            float m2 = red[tid * 8];
#pragma unroll
            for (int j = 1; j < 8; ++j) m2 = fmaxf(m2, red[tid * 8 + j]);
            redF[tid] = m2;
        }
        __syncthreads();
        const float rm = redF[r];
        float sum = 0.f;
        for (int c = l8; c < 512; c += 8) {
            const float e = __expf(S[r * S_PITCH + c] - rm);
            S[r * S_PITCH + c] = e;
            sum += e;
        }
        __syncthreads();
        red[r * 8 + l8] = sum;
        __syncthreads();
        if (tid < 32) {
            float s2 = red[tid * 8];
#pragma unroll
            for (int j = 1; j < 8; ++j) s2 += red[tid * 8 + j];
            redF[tid] = 1.0f / s2;
        }
        __syncthreads();
        const float inv = redF[r];
        for (int c = l8; c < 512; c += 8)
            Ap[r * AP_PITCH + c] = f2bf(S[r * S_PITCH + c] * inv);
        __syncthreads();
    }

    // ---- Phase 3: out GEMM over 2 column chunks of 512 channels.
    // Wave tile 32x64; B tile 512x32 per buffer.
    auto stage3 = [&](int cc, int kc, int buf) {
        const int c0 = cc * 512, k0 = kc * 32;
        unsigned short* Bb = BsBuf(buf);
#pragma unroll
        for (int it = 0; it < 8; ++it) {            // 512 rows x 4 chunks
            const int ch = tid + it * 256;
            const int c = ch >> 2, kp = ch & 3;
            stage16(Bb + c * LP + kp * 8,
                    vc + ((size_t)b * CCH + c0 + c) * NTOK + k0 + kp * 8);
        }
    };

    for (int cc = 0; cc < 2; ++cc) {
        const int c0 = cc * 512;
        f32x8 accO[2][4];
#pragma unroll
        for (int i = 0; i < 2; ++i)
#pragma unroll
            for (int j = 0; j < 4; ++j) accO[i][j] = zero8;

        stage3(cc, 0, 0);
        for (int kc = 0; kc < 16; ++kc) {
            const int buf = kc & 1;
            const int k0 = kc * 32;
            async_fence();
            __syncthreads();
            if (kc + 1 < 16) stage3(cc, kc + 1, buf ^ 1);

            const unsigned short* Bb = BsBuf(buf);
            bf16x16 aF[2], bF[4];
#pragma unroll
            for (int i = 0; i < 2; ++i) {
                const unsigned short* p =
                    Ap + (i * 16 + l15) * AP_PITCH + k0 + 8 * hl;
                aF[i] = lds_frag(p, p + 16);
            }
#pragma unroll
            for (int j = 0; j < 4; ++j) {
                const unsigned short* p =
                    Bb + (wid * 64 + j * 16 + l15) * LP + 8 * hl;
                bF[j] = lds_frag(p, p + 16);
            }
#pragma unroll
            for (int i = 0; i < 2; ++i)
#pragma unroll
                for (int j = 0; j < 4; ++j)
                    accO[i][j] = wmma_bf16(aF[i], bF[j], accO[i][j]);
        }
        // epilogue: transpose-back + residual
#pragma unroll
        for (int j = 0; j < 4; ++j) {
            const int cG = c0 + wid * 64 + j * 16 + l15;
#pragma unroll
            for (int i = 0; i < 2; ++i) {
                const int tokBase = n0 + i * 16 + 8 * hl;
#pragma unroll
                for (int r = 0; r < 8; ++r) {
                    const size_t idx =
                        ((size_t)b * CCH + cG) * NTOK + tokBase + r;
                    out[idx] = accO[i][j][r] + x[idx];
                }
            }
        }
    }
}

// ---------------------------------------------------------------------------
// Host launch
// ---------------------------------------------------------------------------
extern "C" void kernel_launch(void* const* d_in, const int* in_sizes, int n_in,
                              void* d_out, int out_size, void* d_ws,
                              size_t ws_size, hipStream_t stream) {
    const float* x  = (const float*)d_in[0];
    const float* y  = (const float*)d_in[1];
    const float* wq = (const float*)d_in[2];
    const float* bq = (const float*)d_in[3];
    const float* wk = (const float*)d_in[4];
    const float* bk = (const float*)d_in[5];
    const float* wv = (const float*)d_in[6];
    const float* bv = (const float*)d_in[7];
    float* out = (float*)d_out;

    // workspace layout (bytes)
    const size_t WPK_B = (size_t)27 * CCH * CCH * 2;     // 56,623,104
    const size_t ACT_B = (size_t)BATCH * NTOK * CCH * 2; // 16,777,216
    char* ws = (char*)d_ws;
    unsigned short* wpkQ = (unsigned short*)(ws);
    unsigned short* wpkK = (unsigned short*)(ws + WPK_B);
    unsigned short* wpkV = (unsigned short*)(ws + 2 * WPK_B);
    unsigned short* xt   = (unsigned short*)(ws + 3 * WPK_B);
    unsigned short* yt   = (unsigned short*)(ws + 3 * WPK_B + ACT_B);
    unsigned short* qt   = (unsigned short*)(ws + 3 * WPK_B + 2 * ACT_B);
    unsigned short* kt   = (unsigned short*)(ws + 3 * WPK_B + 3 * ACT_B);
    unsigned short* vcm  = (unsigned short*)(ws + 3 * WPK_B + 4 * ACT_B);

    (void)hipFuncSetAttribute((const void*)attention_kernel,
                              hipFuncAttributeMaxDynamicSharedMemorySize,
                              (int)ATT_SMEM);

    // 1. pack weights: 27*2^20 elems / 256 = 110592 blocks, 3 tensors
    pack_weights_kernel<<<dim3(110592, 3), 256, 0, stream>>>(wq, wk, wv, wpkQ,
                                                             wpkK, wpkV);
    // 2. transpose activations: 16 n-tiles x 32 c-tiles x (16 b * 2 tensors)
    transpose_in_kernel<<<dim3(16, 32, 32), 256, 0, stream>>>(x, y, xt, yt);
    // 3. three convs as GEMMs: 4 N-tiles x 8 M-tiles(depth) x 16 batches
    conv_gemm_kernel<<<dim3(4, 8, 16), 256, 0, stream>>>(xt, wpkQ, bq, qt, 0);
    conv_gemm_kernel<<<dim3(4, 8, 16), 256, 0, stream>>>(yt, wpkK, bk, kt, 0);
    conv_gemm_kernel<<<dim3(4, 8, 16), 256, 0, stream>>>(yt, wpkV, bv, vcm, 1);
    // 4. attention: 16 token tiles x 16 batches
    attention_kernel<<<dim3(16, 16), 256, ATT_SMEM, stream>>>(qt, kt, vcm, x,
                                                              out);
}
